// SelfAttention_7679401525317
// MI455X (gfx1250) — matrix operand order repaired and checked by reading, not compiled, not run on hardware
//
#include <hip/hip_runtime.h>
#include <hip/hip_bf16.h>

// ---------------------------------------------------------------------------
// CDNA5 (gfx1250) causal self-attention with 2D-RoPE, bf16 WMMA everywhere.
//   B=2, T=2048, C=768, H=12, D=64
// Pipeline:
//   1) qkv = x @ W_attn                (WMMA GEMM, f32 -> bf16 out)
//   2) RoPE2D(q,k) + relayout          (q,k -> [B,H,T,D] bf16 ; v -> [B,H,D,T] bf16)
//   3) flash attention per (b,h)       (async K/V tiles -> LDS, WMMA QK^T,
//                                       online softmax, WMMA PV)
//   4) y = attn @ W_proj               (WMMA GEMM, bf16 A, f32 out)
// ---------------------------------------------------------------------------

typedef __attribute__((ext_vector_type(16))) __bf16 v16bf;
typedef __attribute__((ext_vector_type(8)))  float  v8f;
typedef __attribute__((__vector_size__(16))) int    vi4;   // int4 vector for async builtins

#define CB  2
#define CT  2048
#define CC  768
#define CH  12
#define CD  64
#define C3C 2304

#define GLOBAL_AS __attribute__((address_space(1)))
#define LDS_AS    __attribute__((address_space(3)))

#if defined(__AMDGCN__) && \
    __has_builtin(__builtin_amdgcn_global_load_async_to_lds_b128) && \
    __has_builtin(__builtin_amdgcn_s_wait_asynccnt)
#define HAS_ASYNC 1
#else
#define HAS_ASYNC 0
#endif

__device__ __forceinline__ unsigned short f2bf(float f) {
  union { float f; unsigned int u; } c; c.f = f;
  unsigned int u = c.u;
  u += 0x7fffu + ((u >> 16) & 1u);       // round-to-nearest-even
  return (unsigned short)(u >> 16);
}
__device__ __forceinline__ float bf2f(unsigned short h) {
  union { unsigned int u; float f; } c; c.u = ((unsigned int)h) << 16;
  return c.f;
}

// Packed f32x2 -> bf16x2 store (v_cvt_pk_bf16_f32 when available).
#if defined(__AMDGCN__) && __has_builtin(__builtin_amdgcn_cvt_pk_bf16_f32)
typedef __attribute__((ext_vector_type(2))) __bf16 v2bf;
__device__ __forceinline__ void store_bf2(unsigned short* p, float a, float b) {
  v2bf r = __builtin_amdgcn_cvt_pk_bf16_f32(a, b);
  *(unsigned int*)p = __builtin_bit_cast(unsigned int, r);
}
#else
__device__ __forceinline__ void store_bf2(unsigned short* p, float a, float b) {
  p[0] = f2bf(a); p[1] = f2bf(b);
}
#endif

// ---------------------------------------------------------------------------
// Tiled GEMM:  out[M,N] = A[M,K] (f32 or bf16) * Bm[K,N] (f32), bf16 WMMA.
// 128x128 tile / workgroup, 256 threads = 8 waves in a 2x4 grid,
// each wave owns 64x32 = 4x2 WMMA tiles. K stepped by 32.
// ---------------------------------------------------------------------------
#define GTM 128
#define GTN 128
#define GKC 32
#define APAD 40   // 80B row stride: 16 lanes land on disjoint 4-bank groups

template<bool A_BF16, bool OUT_BF16>
__global__ __launch_bounds__(256) void gemm_wmma_k(
    const void* __restrict__ Av, const float* __restrict__ Bm,
    void* __restrict__ Ov, int M, int N, int K)
{
  __shared__ __align__(16) unsigned short As[GTM][APAD];
  __shared__ __align__(16) unsigned short Bs[GTN][APAD];   // transposed: [n][k]

  const int tid  = threadIdx.x;
  const int lane = tid & 31;
  const int wave = tid >> 5;
  const int lc   = lane & 15;          // column-in-tile (B/C), row-in-tile (A)
  const int lh   = lane >> 4;          // half-wave select
  const int mbase = blockIdx.x * GTM;
  const int nbase = blockIdx.y * GTN;
  const int waveM = (wave >> 2) * 64;  // 0 or 64
  const int waveN = (wave & 3)  * 32;  // 0,32,64,96

  const v8f vz = {0.f,0.f,0.f,0.f,0.f,0.f,0.f,0.f};
  v8f acc[4][2];
  #pragma unroll
  for (int mt = 0; mt < 4; ++mt)
    #pragma unroll
    for (int nt = 0; nt < 2; ++nt) acc[mt][nt] = vz;

  for (int k0 = 0; k0 < K; k0 += GKC) {
    // ---- stage A tile (128x32) into LDS as bf16 ----
    if (A_BF16) {
      const unsigned short* A = (const unsigned short*)Av;
      #pragma unroll
      for (int it = 0; it < 4; ++it) {
        int l   = (it * 256 + tid) * 4;              // 0..4095
        int row = l >> 5, kk = l & 31;
        const unsigned short* src = A + (size_t)(mbase + row) * K + k0 + kk;
        As[row][kk+0] = src[0]; As[row][kk+1] = src[1];
        As[row][kk+2] = src[2]; As[row][kk+3] = src[3];
      }
    } else {
      const float* A = (const float*)Av;
      #pragma unroll
      for (int it = 0; it < 4; ++it) {
        int l   = (it * 256 + tid) * 4;
        int row = l >> 5, kk = l & 31;
        float4 g = *(const float4*)(A + (size_t)(mbase + row) * K + k0 + kk);
        store_bf2(&As[row][kk+0], g.x, g.y);
        store_bf2(&As[row][kk+2], g.z, g.w);
      }
    }
    // ---- stage B tile (32x128) transposed into LDS as bf16 ----
    #pragma unroll
    for (int it = 0; it < 4; ++it) {
      int l  = (it * 256 + tid) * 4;                 // 0..4095
      int kr = l >> 7, n = l & 127;
      float4 g = *(const float4*)(Bm + (size_t)(k0 + kr) * N + nbase + n);
      Bs[n+0][kr] = f2bf(g.x); Bs[n+1][kr] = f2bf(g.y);
      Bs[n+2][kr] = f2bf(g.z); Bs[n+3][kr] = f2bf(g.w);
    }
    __syncthreads();

    // ---- fragments + WMMA ----
    v16bf afr[4], bfr[2];
    #pragma unroll
    for (int mt = 0; mt < 4; ++mt) {
      // A 16x32 layout: lane lc = row; K chunks [8*lh..+7] and [16+8*lh..+7]
      const unsigned short* p = &As[waveM + mt*16 + lc][8*lh];
      uint4* d = (uint4*)&afr[mt];
      d[0] = *(const uint4*)p;
      d[1] = *(const uint4*)(p + 16);
    }
    #pragma unroll
    for (int nt = 0; nt < 2; ++nt) {
      // B 32x16 layout: lane lc = col; K chunks [8*lh..+7] and [16+8*lh..+7]
      const unsigned short* p = &Bs[waveN + nt*16 + lc][8*lh];
      uint4* d = (uint4*)&bfr[nt];
      d[0] = *(const uint4*)p;
      d[1] = *(const uint4*)(p + 16);
    }
    #pragma unroll
    for (int mt = 0; mt < 4; ++mt)
      #pragma unroll
      for (int nt = 0; nt < 2; ++nt)
        acc[mt][nt] = __builtin_amdgcn_wmma_f32_16x16x32_bf16(
            false, afr[mt], false, bfr[nt], (short)0, acc[mt][nt], false, false);
    __syncthreads();
  }

  // ---- epilogue: C layout -> global ----
  #pragma unroll
  for (int mt = 0; mt < 4; ++mt) {
    #pragma unroll
    for (int nt = 0; nt < 2; ++nt) {
      #pragma unroll
      for (int r = 0; r < 8; ++r) {
        int row = mbase + waveM + mt*16 + r + 8*lh;
        int col = nbase + waveN + nt*16 + lc;
        float v = acc[mt][nt][r];
        if (OUT_BF16) ((unsigned short*)Ov)[(size_t)row * N + col] = f2bf(v);
        else          ((float*)Ov)[(size_t)row * N + col] = v;
      }
    }
  }
}

// ---------------------------------------------------------------------------
// RoPE-2D + relayout. One block per (b,t).
//   qkv bf16 [B,T,3C]  ->  q,k bf16 [B,H,T,D] (rotated), vT bf16 [B,H,D,T]
// Quarter 0/1 (d 0..31) rotate with pos[...,0]; quarter 2/3 with pos[...,1].
// ---------------------------------------------------------------------------
__global__ __launch_bounds__(256) void rope_prep_k(
    const unsigned short* __restrict__ qkv, const int* __restrict__ pos,
    unsigned short* __restrict__ Qo, unsigned short* __restrict__ Ko,
    unsigned short* __restrict__ VT)
{
  const int bt = blockIdx.x;
  const int b = bt / CT, t = bt % CT;
  const float p0 = (float)pos[bt*2 + 0];
  const float p1 = (float)pos[bt*2 + 1];
  const int tid = threadIdx.x;

  // q and k: 2 tensors * H * 32 rotate-pairs = 768 work items
  for (int idx = tid; idx < 2 * CH * 32; idx += 256) {
    int tensor = idx / (CH * 32);
    int r  = idx % (CH * 32);
    int hh = r / 32, j = r % 32;
    int quarter = j >> 4, i = j & 15;
    float ang  = (quarter ? p1 : p0) * __expf(-(float)i * 0.57564627324851147f); // ln(1e4)/16
    float cs   = __cosf(ang), sn = __sinf(ang);
    size_t bin = (size_t)bt * C3C + (size_t)tensor * CC + hh * CD + quarter * 32;
    float x1 = bf2f(qkv[bin + i]);
    float x2 = bf2f(qkv[bin + 16 + i]);
    unsigned short* out = tensor ? Ko : Qo;
    size_t bo = (((size_t)b * CH + hh) * CT + t) * CD + quarter * 32;
    out[bo + i]      = f2bf(x1 * cs - x2 * sn);
    out[bo + 16 + i] = f2bf(x1 * sn + x2 * cs);
  }
  // v: transpose-copy 768 elements, [B,T,C] -> [B,H,D,T]
  for (int idx = tid; idx < CC; idx += 256) {
    int hh = idx / CD, d = idx % CD;
    VT[(((size_t)b * CH + hh) * CD + d) * CT + t] =
        qkv[(size_t)bt * C3C + 2 * CC + idx];
  }
}

// ---------------------------------------------------------------------------
// Flash attention. Block = 128 threads = 4 waves; wave owns 16 q rows.
// S chunked by 64. All waves in a block share the same chunk sequence
// (trip count qblk+1 is wave-uniform), so K and V^T chunk tiles are staged
// cooperatively into LDS — with GLOBAL_LOAD_ASYNC_TO_LDS_B128 double
// buffering when the toolchain exposes it (ASYNCcnt + barrier ordering).
// P goes through per-wave LDS to convert C-layout -> A-layout as bf16.
// ---------------------------------------------------------------------------
__global__ __launch_bounds__(128) void attn_k(
    const unsigned short* __restrict__ Q,    // [B*H, T, D]
    const unsigned short* __restrict__ Kt,   // [B*H, T, D]
    const unsigned short* __restrict__ Vt,   // [B*H, D, T]
    unsigned short* __restrict__ O)          // [B, T, C] bf16
{
  __shared__ __align__(16) unsigned short Ksh[2][64][72]; // [s][d], padded rows
  __shared__ __align__(16) unsigned short Vsh[2][64][72]; // [d][s], padded rows
  __shared__ __align__(16) unsigned short P[4][16][72];   // per-wave P tile

  const int lane = threadIdx.x & 31;
  const int wave = threadIdx.x >> 5;
  const int lc = lane & 15, lh = lane >> 4;
  const int qblk = blockIdx.x & 31;        // T/64 = 32 q blocks
  const int bh   = blockIdx.x >> 5;        // 0..23
  const int h = bh % CH, b = bh / CH;
  const int rowbase = qblk * 64 + wave * 16;

  const unsigned short* qp = Q  + ((size_t)bh * CT + rowbase) * CD;
  const unsigned short* kp = Kt + (size_t)bh * CT * CD;
  const unsigned short* vp = Vt + (size_t)bh * CD * CT;

  // Cooperative stage of one 64-wide S chunk of K and V^T into LDS.
  // 2 tiles * 64 rows * 128B = 2*8KB; 128 threads x 4 iters x 2x16B.
  auto stage_kv = [&](int buf, int s0) {
    #pragma unroll
    for (int it = 0; it < 4; ++it) {
      int i   = it * 128 + (int)threadIdx.x;   // 0..511
      int row = i >> 3;                        // K: s-local, V: d
      int c   = (i & 7) * 8;                   // 16B-granular column
      const unsigned short* gk = kp + (size_t)(s0 + row) * CD + c;
      const unsigned short* gv = vp + (size_t)row * CT + s0 + c;
#if HAS_ASYNC
      __builtin_amdgcn_global_load_async_to_lds_b128(
          (GLOBAL_AS vi4*)gk, (LDS_AS vi4*)&Ksh[buf][row][c], 0, 0);
      __builtin_amdgcn_global_load_async_to_lds_b128(
          (GLOBAL_AS vi4*)gv, (LDS_AS vi4*)&Vsh[buf][row][c], 0, 0);
#else
      *(uint4*)&Ksh[buf][row][c] = *(const uint4*)gk;
      *(uint4*)&Vsh[buf][row][c] = *(const uint4*)gv;
#endif
    }
  };

  // Q A-fragments for both 32-wide K steps over D=64 (loaded once, global)
  v16bf qf[2];
  #pragma unroll
  for (int ks = 0; ks < 2; ++ks) {
    const unsigned short* p = qp + (size_t)lc * CD + ks * 32 + 8 * lh;
    uint4* d = (uint4*)&qf[ks];
    d[0] = *(const uint4*)p;
    d[1] = *(const uint4*)(p + 16);
  }

  const v8f vz = {0.f,0.f,0.f,0.f,0.f,0.f,0.f,0.f};
  v8f oacc[4];
  #pragma unroll
  for (int jd = 0; jd < 4; ++jd) oacc[jd] = vz;
  float mr[8], lr[8];
  #pragma unroll
  for (int r = 0; r < 8; ++r) { mr[r] = -3.0e38f; lr[r] = 0.f; }
  const float scale = 0.125f;   // 1/sqrt(64)

  const int nchunk = qblk + 1;  // uniform across all waves in the block

  stage_kv(0, 0);
#if HAS_ASYNC
  __builtin_amdgcn_s_wait_asynccnt(0);
#endif
  __syncthreads();

  for (int cidx = 0; cidx < nchunk; ++cidx) {
    const int s0  = cidx * 64;
    const int cur = cidx & 1;

    // prefetch next chunk while computing this one
    if (cidx + 1 < nchunk) stage_kv(cur ^ 1, s0 + 64);

    // ---- S = Q K^T for 4 column tiles ----
    v8f sacc[4];
    #pragma unroll
    for (int j = 0; j < 4; ++j) sacc[j] = vz;
    #pragma unroll
    for (int ks = 0; ks < 2; ++ks) {
      #pragma unroll
      for (int j = 0; j < 4; ++j) {
        const unsigned short* p = &Ksh[cur][j*16 + lc][ks*32 + lh*8];
        v16bf kf; uint4* d = (uint4*)&kf;
        d[0] = *(const uint4*)p;
        d[1] = *(const uint4*)(p + 16);
        sacc[j] = __builtin_amdgcn_wmma_f32_16x16x32_bf16(
            false, qf[ks], false, kf, (short)0, sacc[j], false, false);
      }
    }

    // ---- scale + causal mask ----
    #pragma unroll
    for (int j = 0; j < 4; ++j) {
      int col = s0 + j*16 + lc;
      #pragma unroll
      for (int r = 0; r < 8; ++r) {
        int row = rowbase + r + 8*lh;
        float v = sacc[j][r] * scale;
        sacc[j][r] = (col <= row) ? v : -3.0e38f;
      }
    }

    // ---- online softmax (row spans 16 lanes of one half-wave) ----
    #pragma unroll
    for (int r = 0; r < 8; ++r) {
      float v = fmaxf(fmaxf(sacc[0][r], sacc[1][r]), fmaxf(sacc[2][r], sacc[3][r]));
      v = fmaxf(v, __shfl_xor(v, 1)); v = fmaxf(v, __shfl_xor(v, 2));
      v = fmaxf(v, __shfl_xor(v, 4)); v = fmaxf(v, __shfl_xor(v, 8));
      float mn = fmaxf(mr[r], v);
      float alpha = __expf(mr[r] - mn);
      mr[r] = mn;
      float ps = 0.f;
      #pragma unroll
      for (int j = 0; j < 4; ++j) {
        float pv = __expf(sacc[j][r] - mn);
        ps += pv;
        P[wave][r + 8*lh][j*16 + lc] = f2bf(pv);
      }
      ps += __shfl_xor(ps, 1); ps += __shfl_xor(ps, 2);
      ps += __shfl_xor(ps, 4); ps += __shfl_xor(ps, 8);
      lr[r] = lr[r] * alpha + ps;
      #pragma unroll
      for (int jd = 0; jd < 4; ++jd) oacc[jd][r] *= alpha;
    }

    // ---- O += P V  (P from LDS in A-layout, V^T tile from LDS) ----
    #pragma unroll
    for (int ks = 0; ks < 2; ++ks) {
      const unsigned short* pp = &P[wave][lc][ks * 32 + 8 * lh];
      v16bf af; uint4* d = (uint4*)&af;
      d[0] = *(const uint4*)pp;
      d[1] = *(const uint4*)(pp + 16);
      #pragma unroll
      for (int jd = 0; jd < 4; ++jd) {
        const unsigned short* vpt = &Vsh[cur][jd*16 + lc][ks*32 + lh*8];
        v16bf vf; uint4* dv = (uint4*)&vf;
        dv[0] = *(const uint4*)vpt;
        dv[1] = *(const uint4*)(vpt + 16);
        oacc[jd] = __builtin_amdgcn_wmma_f32_16x16x32_bf16(
            false, af, false, vf, (short)0, oacc[jd], false, false);
      }
    }

    if (cidx + 1 < nchunk) {
#if HAS_ASYNC
      __builtin_amdgcn_s_wait_asynccnt(0);
#endif
      __syncthreads();
    }
  }

  // ---- epilogue: normalize and scatter to [B,T,C] bf16 ----
  #pragma unroll
  for (int jd = 0; jd < 4; ++jd) {
    #pragma unroll
    for (int r = 0; r < 8; ++r) {
      int t = rowbase + r + 8*lh;
      int c = h * CD + jd*16 + lc;
      O[((size_t)b * CT + t) * CC + c] = f2bf(oacc[jd][r] / lr[r]);
    }
  }
}

// ---------------------------------------------------------------------------
// Host-side launch
// ---------------------------------------------------------------------------
extern "C" void kernel_launch(void* const* d_in, const int* in_sizes, int n_in,
                              void* d_out, int out_size, void* d_ws, size_t ws_size,
                              hipStream_t stream) {
  (void)in_sizes; (void)n_in; (void)out_size; (void)ws_size;
  const float* x      = (const float*)d_in[0];  // [B,T,C]
  const float* W_attn = (const float*)d_in[1];  // [C,3C]
  const float* W_proj = (const float*)d_in[2];  // [C,C]
  const int*   pos    = (const int*)d_in[4];    // [B,T,2]  (d_in[3] mask: causal, computed)

  const size_t nBT   = (size_t)CB * CT;            // 4096
  unsigned short* qkv_bf  = (unsigned short*)d_ws;                 // [B,T,3C]
  unsigned short* q_bf    = qkv_bf + nBT * C3C;                    // [B,H,T,D]
  unsigned short* k_bf    = q_bf   + nBT * CC;                     // [B,H,T,D]
  unsigned short* vT_bf   = k_bf   + nBT * CC;                     // [B,H,D,T]
  unsigned short* attn_bf = vT_bf  + nBT * CC;                     // [B,T,C]

  // 1) qkv = x @ W_attn  (f32 in, bf16 out)
  gemm_wmma_k<false, true><<<dim3(nBT / GTM, C3C / GTN), 256, 0, stream>>>(
      (const void*)x, W_attn, (void*)qkv_bf, (int)nBT, C3C, CC);

  // 2) RoPE-2D + relayout
  rope_prep_k<<<dim3((unsigned)nBT), 256, 0, stream>>>(qkv_bf, pos, q_bf, k_bf, vT_bf);

  // 3) causal flash attention
  attn_k<<<dim3(CB * CH * (CT / 64)), 128, 0, stream>>>(q_bf, k_bf, vT_bf, attn_bf);

  // 4) y = attn @ W_proj  (bf16 A, f32 out)
  gemm_wmma_k<true, false><<<dim3(nBT / GTM, CC / GTN), 256, 0, stream>>>(
      (const void*)attn_bf, W_proj, d_out, (int)nBT, CC, CC);
}
